// Conv2d_19851338842267
// MI455X (gfx1250) — matrix operand order repaired and checked by reading, not compiled, not run on hardware
//
#include <hip/hip_runtime.h>
#include <hip/hip_bf16.h>

// GEMM: out[64,12544] = X[64,6272] * W[12544,6272]^T + bias[12544]
// M=64, N=12544, K=6272, f32 in memory.
//
// Roofline: weight matrix = 315 MB > 192 MB L2 -> streams from HBM each call:
// floor = 315 MB / 23.3 TB/s ~= 13.5 us. bf16 WMMA compute ~= 4 us, so the
// right design is: stream f32 weights exactly once, convert to bf16 in-register,
// V_WMMA_F32_16X16X32_BF16 with f32 accumulation.
//
// Kernel 1: convert enc_x (1.6 MB) to bf16 in d_ws once per call (~0.1 us),
//           so the GEMM loads A fragments directly in WMMA register layout.
// Kernel 2: GEMM, one wave owns 64(M) x 16(N), ping-pong software pipeline.

typedef __attribute__((ext_vector_type(16))) __bf16 v16bf;
typedef __attribute__((ext_vector_type(8)))  __bf16 v8bf;
typedef __attribute__((ext_vector_type(4)))  __bf16 v4bf;
typedef __attribute__((ext_vector_type(8)))  float  v8f;

namespace {
constexpr int KD = 6272;   // IC*IH*IW
constexpr int ND = 12544;  // OC*OH*OW
constexpr int MD = 64;     // BATCH
}

// ---------------------------------------------------------------------------
// Kernel 1: X f32 [64,6272] -> bf16 [64,6272] in d_ws.
// 64*6272/4 = 100352 float4 chunks = 392 blocks * 256 threads exactly.
// ---------------------------------------------------------------------------
__global__ __launch_bounds__(256)
void cvt_x_to_bf16(const float* __restrict__ X, v4bf* __restrict__ Y) {
  const int i = blockIdx.x * 256 + threadIdx.x;  // exact grid, no tail
  float4 v = ((const float4*)X)[i];
  v4bf o;
  o[0] = (__bf16)v.x; o[1] = (__bf16)v.y; o[2] = (__bf16)v.z; o[3] = (__bf16)v.w;
  Y[i] = o;
}

// ---------------------------------------------------------------------------
// Kernel 2: GEMM with bias, bf16 WMMA, ping-pong pipelined K loop.
// ---------------------------------------------------------------------------
struct Stage {
  float4 w[4];     // 16 consecutive f32 of this lane's weight row
  v8bf   a[4][2];  // A fragments (already bf16, already in WMMA layout)
};

static __device__ __forceinline__ void load_stage(Stage& s,
                                                  const float* __restrict__ wp,
                                                  const __bf16* __restrict__ xp,
                                                  int k0) {
  // Prefetch the weight stream ~16 iterations (2 KB/lane) ahead.
  __builtin_prefetch(wp + k0 + 512, 0, 0);
  s.w[0] = *(const float4*)(wp + k0);
  s.w[1] = *(const float4*)(wp + k0 + 4);
  s.w[2] = *(const float4*)(wp + k0 + 8);
  s.w[3] = *(const float4*)(wp + k0 + 12);
#pragma unroll
  for (int t = 0; t < 4; ++t) {
    const __bf16* xt = xp + (size_t)(t * 16) * KD + k0;
    s.a[t][0] = *(const v8bf*)(xt);        // K = kA .. kA+7
    s.a[t][1] = *(const v8bf*)(xt + 8);    // K = kA+8 .. kA+15
  }
}

static __device__ __forceinline__ v16bf pack_b(const Stage& s) {
  v16bf r;
  r[0]  = (__bf16)s.w[0].x; r[1]  = (__bf16)s.w[0].y;
  r[2]  = (__bf16)s.w[0].z; r[3]  = (__bf16)s.w[0].w;
  r[4]  = (__bf16)s.w[1].x; r[5]  = (__bf16)s.w[1].y;
  r[6]  = (__bf16)s.w[1].z; r[7]  = (__bf16)s.w[1].w;
  r[8]  = (__bf16)s.w[2].x; r[9]  = (__bf16)s.w[2].y;
  r[10] = (__bf16)s.w[2].z; r[11] = (__bf16)s.w[2].w;
  r[12] = (__bf16)s.w[3].x; r[13] = (__bf16)s.w[3].y;
  r[14] = (__bf16)s.w[3].z; r[15] = (__bf16)s.w[3].w;
  return r;
}

static __device__ __forceinline__ void compute_stage(const Stage& s, v8f acc[4]) {
  v16bf bm = pack_b(s);
#pragma unroll
  for (int t = 0; t < 4; ++t) {
    v16bf am = __builtin_shufflevector(s.a[t][0], s.a[t][1],
                                       0, 1, 2, 3, 4, 5, 6, 7,
                                       8, 9, 10, 11, 12, 13, 14, 15);
    // (neg_a, A, neg_b, B, c_mod, C, reuse_a, reuse_b)
    acc[t] = __builtin_amdgcn_wmma_f32_16x16x32_bf16(
        false, am, false, bm, (short)0, acc[t], false, false);
  }
}

__global__ __launch_bounds__(128, 1)
void conv_toeplitz_gemm_wmma(const __bf16* __restrict__ Xb,
                             const float* __restrict__ W,
                             const float* __restrict__ Bv,
                             float* __restrict__ O) {
  const int lane = threadIdx.x & 31;   // wave32
  const int wave = threadIdx.x >> 5;   // 4 waves per block
  const int sub  = lane & 15;          // N column within tile / M row within tile
  const int hi   = lane >> 4;          // half-wave selector
  const int col  = blockIdx.x * 64 + wave * 16 + sub;  // global N column

  // Accumulators: 4 M-tiles of 16x16 f32; C/D layout: VGPR r, lane l ->
  // row = r + (l<16 ? 0 : 8), col = l&15. Bias depends only on the column:
  // pre-load it into C so the epilogue is pure stores.
  const float bias = Bv[col];
  v8f acc[4];
#pragma unroll
  for (int t = 0; t < 4; ++t) {
#pragma unroll
    for (int i = 0; i < 8; ++i) acc[t][i] = bias;
  }

  // B operand (32K x 16N bf16): lane = column n, VGPR v holds K=2v,2v+1
  // (+16 for lanes 16..31) -> 16 consecutive f32 of weight row `col`.
  const float* wp = W + (size_t)col * KD + hi * 16;
  // A operand (16M x 32K bf16): lanes 0..15 row M=lane hold K k0+0..15;
  // lanes 16..31 row M=lane-16 hold K k0+16..31.
  const __bf16* xp = Xb + (size_t)sub * KD + hi * 16;

  // Ping-pong software pipeline over K (KD/32 = 196 steps, even).
  Stage s0, s1;
  load_stage(s0, wp, xp, 0);
  int k0 = 0;
  for (; k0 + 64 < KD; k0 += 64) {
    load_stage(s1, wp, xp, k0 + 32);
    compute_stage(s0, acc);
    load_stage(s0, wp, xp, k0 + 64);
    compute_stage(s1, acc);
  }
  // k0 == KD-64 here: one stage in flight, one chunk left to load.
  load_stage(s1, wp, xp, k0 + 32);
  compute_stage(s0, acc);
  compute_stage(s1, acc);

  // Store: row = t*16 + r + hi*8, column `col`; out row-major [64, 12544].
#pragma unroll
  for (int t = 0; t < 4; ++t) {
#pragma unroll
    for (int r = 0; r < 8; ++r) {
      O[(size_t)(t * 16 + r + hi * 8) * ND + col] = acc[t][r];
    }
  }
}

extern "C" void kernel_launch(void* const* d_in, const int* in_sizes, int n_in,
                              void* d_out, int out_size, void* d_ws, size_t ws_size,
                              hipStream_t stream) {
  (void)in_sizes; (void)n_in; (void)out_size; (void)ws_size;
  const float* x = (const float*)d_in[0];  // enc_x  [64, 6272]
  const float* w = (const float*)d_in[1];  // weight [12544, 6272]
  const float* b = (const float*)d_in[2];  // bias   [12544]
  float* o = (float*)d_out;                // out    [64, 12544]

  // Stage 1: enc_x -> bf16 scratch (64*6272*2 = 802,816 bytes of d_ws).
  v4bf* xb = (v4bf*)d_ws;
  cvt_x_to_bf16<<<dim3((MD * KD) / (4 * 256)), dim3(256), 0, stream>>>(x, xb);

  // Stage 2: GEMM. 4 waves/block, each wave owns 16 N columns; 196 blocks
  // -> 784 waves; weight matrix is read exactly once.
  conv_toeplitz_gemm_wmma<<<dim3(ND / 64), dim3(128), 0, stream>>>(
      (const __bf16*)d_ws, w, b, o);
}
